// MoEExperts_35098472742973
// MI455X (gfx1250) — matrix operand order repaired and checked by reading, not compiled, not run on hardware
//
#include <hip/hip_runtime.h>
#include <math.h>

#define NUM_EXPERTS 8
#define HIDDEN 2048
#define INTER 4096
#define TOKENS 4096
#define TOPK 2
#define NENT (TOKENS * TOPK)   // 8192 routed (token, slot) entries

#define NT 512   // threads per GEMM block = 16 waves
#define TM 256   // M tile (entries)  -> halves weight re-read traffic vs 128
#define TN 64    // N tile
#define KC 64    // K chunk per LDS stage (= 2 WMMA K-steps)
#define KCP (KC + 8)  // padded LDS stride (144 B, 16B-aligned)

typedef __attribute__((ext_vector_type(16))) __bf16 v16bf;
typedef __attribute__((ext_vector_type(8)))  float  v8f;
typedef int v4i __attribute__((vector_size(16)));

struct alignas(16) U4 { unsigned x, y, z, w; };
union ABf { U4 q[2]; v16bf v; };

#if __has_builtin(__builtin_amdgcn_global_load_async_to_lds_b128)
#define HAVE_ASYNC_LDS 1
#endif

__device__ __forceinline__ void async_wait0() {
#ifdef HAVE_ASYNC_LDS
#if __has_builtin(__builtin_amdgcn_s_wait_asynccnt)
  __builtin_amdgcn_s_wait_asynccnt(0);
#else
  asm volatile("s_wait_asynccnt 0" ::: "memory");
#endif
#endif
}

__device__ __forceinline__ void copy16_to_lds(const unsigned short* src, unsigned short* ldsDst) {
#ifdef HAVE_ASYNC_LDS
  __builtin_amdgcn_global_load_async_to_lds_b128(
      (__attribute__((address_space(1))) v4i*)src,
      (__attribute__((address_space(3))) v4i*)ldsDst, 0, 0);
#else
  *(uint4*)ldsDst = *(const uint4*)src;
#endif
}

__device__ __forceinline__ unsigned short f2bf(float f) {
  union { float f; unsigned u; } c; c.f = f;
  unsigned u = c.u;
  return (unsigned short)((u + 0x7FFFu + ((u >> 16) & 1u)) >> 16);  // RNE
}

// ---------------- small utility kernels ----------------

__global__ void k_zero_out(float4* __restrict__ p, int n4) {
  int i = blockIdx.x * blockDim.x + threadIdx.x;
  if (i < n4) { float4 z; z.x = z.y = z.z = z.w = 0.f; p[i] = z; }
}

__global__ void k_init_meta(int* __restrict__ counts, int* __restrict__ zeroPad) {
  int t = threadIdx.x;
  if (t < NUM_EXPERTS) counts[t] = 0;
  if (t < 64) zeroPad[t] = 0;   // 256 B zero block for OOB async reads
}

__global__ void k_count(const int* __restrict__ eidx, int* __restrict__ counts) {
  int i = blockIdx.x * blockDim.x + threadIdx.x;
  if (i < NENT) atomicAdd(&counts[eidx[i]], 1);
}

__global__ void k_scan(const int* __restrict__ counts, int* __restrict__ offs,
                       int* __restrict__ cursors) {
  offs[0] = 0;
  for (int e = 0; e < NUM_EXPERTS; ++e) {
    offs[e + 1] = offs[e] + counts[e];
    cursors[e] = offs[e];
  }
}

__global__ void k_fill(const int* __restrict__ eidx, const float* __restrict__ ew,
                       int* __restrict__ cursors, int* __restrict__ tokenIds,
                       float* __restrict__ wts) {
  int i = blockIdx.x * blockDim.x + threadIdx.x;
  if (i < NENT) {
    int e = eidx[i];
    int pos = atomicAdd(&cursors[e], 1);
    tokenIds[pos] = i >> 1;       // token index
    wts[pos] = ew[i];             // routing weight for this slot
  }
}

__global__ void k_cvt_x(const float* __restrict__ x, unsigned short* __restrict__ xbf) {
  int i = blockIdx.x * blockDim.x + threadIdx.x;   // one float4 per thread
  float4 v = ((const float4*)x)[i];
  unsigned lo = (unsigned)f2bf(v.x) | ((unsigned)f2bf(v.y) << 16);
  unsigned hi = (unsigned)f2bf(v.z) | ((unsigned)f2bf(v.w) << 16);
  uint2 o; o.x = lo; o.y = hi;
  ((uint2*)xbf)[i] = o;
}

// ---------------- Stage A: h = silu(X·W1) * (X·W3), per-expert gathered rows ----------------

__global__ __launch_bounds__(NT) void k_gate_up(
    const unsigned short* __restrict__ xbf,
    const float* __restrict__ w1s, const float* __restrict__ w3s,
    const int* __restrict__ tokenIds, const int* __restrict__ offs,
    const unsigned short* __restrict__ zeroPad,
    unsigned short* __restrict__ hbuf)
{
  __shared__ unsigned short As[TM][KCP];     // 36 KB
  __shared__ unsigned short B1s[TN][KCP];    // 9 KB
  __shared__ unsigned short B3s[TN][KCP];    // 9 KB

  const int e = blockIdx.z;
  const int rowStart = offs[e];
  const int cnt = offs[e + 1] - rowStart;
  const int m0 = blockIdx.y * TM;
  if (m0 >= cnt) return;                   // block-uniform early exit
  const int n0 = blockIdx.x * TN;

  const int tid = threadIdx.x;
  const int lane = tid & 31;
  const int wave = tid >> 5;               // 0..15
  const int wm = wave >> 1;                // 0..7 -> 32-row band
  const int wn = wave & 1;                 // 0..1 -> 32-col band
  const int lhi = lane >> 4;               // lane half
  const int llo = lane & 15;

  const float* __restrict__ W1 = w1s + (size_t)e * HIDDEN * INTER;
  const float* __restrict__ W3 = w3s + (size_t)e * HIDDEN * INTER;

  const v8f vzero = {0.f, 0.f, 0.f, 0.f, 0.f, 0.f, 0.f, 0.f};
  v8f accg[2][2], accu[2][2];
  for (int i = 0; i < 2; ++i)
    for (int j = 0; j < 2; ++j) { accg[i][j] = vzero; accu[i][j] = vzero; }

  for (int k0 = 0; k0 < HIDDEN; k0 += KC) {
    // A tile: 256 x 64 bf16 (gathered token rows), async copy to LDS
    #pragma unroll
    for (int c = 0; c < 4; ++c) {
      int chunk = tid + c * NT;            // 2048 16B-chunks
      int r = chunk >> 3;
      int cc = (chunk & 7) << 3;
      const unsigned short* src = zeroPad;
      if (m0 + r < cnt) {
        int tok = tokenIds[rowStart + m0 + r];
        src = xbf + (size_t)tok * HIDDEN + k0 + cc;
      }
      copy16_to_lds(src, &As[r][cc]);
    }
    // B tiles: 64(K) x 64(N) fp32 -> bf16, transposed store [N][K]
    #pragma unroll
    for (int c = 0; c < 2; ++c) {
      int q = tid + c * NT;                // 1024 float4s per matrix
      int kk = q >> 4;
      int nn = (q & 15) << 2;
      const float4 a1 = *(const float4*)(W1 + (size_t)(k0 + kk) * INTER + n0 + nn);
      const float4 a3 = *(const float4*)(W3 + (size_t)(k0 + kk) * INTER + n0 + nn);
      if (k0 + KC < HIDDEN) {              // prefetch next K-chunk of weights
        __builtin_prefetch(W1 + (size_t)(k0 + KC + kk) * INTER + n0 + nn, 0, 1);
        __builtin_prefetch(W3 + (size_t)(k0 + KC + kk) * INTER + n0 + nn, 0, 1);
      }
      B1s[nn + 0][kk] = f2bf(a1.x); B1s[nn + 1][kk] = f2bf(a1.y);
      B1s[nn + 2][kk] = f2bf(a1.z); B1s[nn + 3][kk] = f2bf(a1.w);
      B3s[nn + 0][kk] = f2bf(a3.x); B3s[nn + 1][kk] = f2bf(a3.y);
      B3s[nn + 2][kk] = f2bf(a3.z); B3s[nn + 3][kk] = f2bf(a3.w);
    }
    async_wait0();
    __syncthreads();

    #pragma unroll
    for (int ks = 0; ks < KC; ks += 32) {
      ABf a[2], b1[2], b3[2];
      #pragma unroll
      for (int mi = 0; mi < 2; ++mi) {     // A frag: row=llo, K {0..15} / {16..31}
        int r = wm * 32 + mi * 16 + llo;
        int c0 = ks + (lhi << 4);
        a[mi].q[0] = *(const U4*)&As[r][c0];
        a[mi].q[1] = *(const U4*)&As[r][c0 + 8];
      }
      #pragma unroll
      for (int ni = 0; ni < 2; ++ni) {     // B frag: col=llo, K half by lane half
        int n = wn * 32 + ni * 16 + llo;
        int kb = ks + (lhi << 4);
        b1[ni].q[0] = *(const U4*)&B1s[n][kb];
        b1[ni].q[1] = *(const U4*)&B1s[n][kb + 8];
        b3[ni].q[0] = *(const U4*)&B3s[n][kb];
        b3[ni].q[1] = *(const U4*)&B3s[n][kb + 8];
      }
      #pragma unroll
      for (int mi = 0; mi < 2; ++mi)
        #pragma unroll
        for (int ni = 0; ni < 2; ++ni) {
          accg[mi][ni] = __builtin_amdgcn_wmma_f32_16x16x32_bf16(
              false, a[mi].v, false, b1[ni].v, (short)0, accg[mi][ni], false, false);
          accu[mi][ni] = __builtin_amdgcn_wmma_f32_16x16x32_bf16(
              false, a[mi].v, false, b3[ni].v, (short)0, accu[mi][ni], false, false);
        }
    }
    __syncthreads();
  }

  // epilogue: h = silu(g) * u  (C/D layout: row = 8*lhi + vgpr, col = llo)
  #pragma unroll
  for (int mi = 0; mi < 2; ++mi)
    #pragma unroll
    for (int ni = 0; ni < 2; ++ni)
      #pragma unroll
      for (int j = 0; j < 8; ++j) {
        int r = wm * 32 + mi * 16 + (lhi << 3) + j;
        if (m0 + r < cnt) {
          int col = n0 + wn * 32 + ni * 16 + llo;
          float g = accg[mi][ni][j];
          float u = accu[mi][ni][j];
          float s = g / (1.0f + __expf(-g));
          hbuf[(size_t)(rowStart + m0 + r) * INTER + col] = f2bf(s * u);
        }
      }
}

// ---------------- Stage B: out[tok] += w * (h · W2) ----------------

__global__ __launch_bounds__(NT) void k_down(
    const unsigned short* __restrict__ hbuf,
    const float* __restrict__ w2s,
    const int* __restrict__ tokenIds, const float* __restrict__ wts,
    const int* __restrict__ offs,
    const unsigned short* __restrict__ zeroPad,
    float* __restrict__ out)
{
  __shared__ unsigned short As[TM][KCP];
  __shared__ unsigned short Bs[TN][KCP];

  const int e = blockIdx.z;
  const int rowStart = offs[e];
  const int cnt = offs[e + 1] - rowStart;
  const int m0 = blockIdx.y * TM;
  if (m0 >= cnt) return;
  const int n0 = blockIdx.x * TN;

  const int tid = threadIdx.x;
  const int lane = tid & 31;
  const int wave = tid >> 5;
  const int wm = wave >> 1;
  const int wn = wave & 1;
  const int lhi = lane >> 4;
  const int llo = lane & 15;

  const float* __restrict__ W2 = w2s + (size_t)e * INTER * HIDDEN;

  const v8f vzero = {0.f, 0.f, 0.f, 0.f, 0.f, 0.f, 0.f, 0.f};
  v8f acc[2][2];
  for (int i = 0; i < 2; ++i)
    for (int j = 0; j < 2; ++j) acc[i][j] = vzero;

  for (int k0 = 0; k0 < INTER; k0 += KC) {
    #pragma unroll
    for (int c = 0; c < 4; ++c) {
      int chunk = tid + c * NT;
      int r = chunk >> 3;
      int cc = (chunk & 7) << 3;
      const unsigned short* src = zeroPad;
      if (m0 + r < cnt)
        src = hbuf + (size_t)(rowStart + m0 + r) * INTER + k0 + cc;
      copy16_to_lds(src, &As[r][cc]);
    }
    #pragma unroll
    for (int c = 0; c < 2; ++c) {
      int q = tid + c * NT;
      int kk = q >> 4;
      int nn = (q & 15) << 2;
      const float4 a2 = *(const float4*)(W2 + (size_t)(k0 + kk) * HIDDEN + n0 + nn);
      if (k0 + KC < INTER)
        __builtin_prefetch(W2 + (size_t)(k0 + KC + kk) * HIDDEN + n0 + nn, 0, 1);
      Bs[nn + 0][kk] = f2bf(a2.x); Bs[nn + 1][kk] = f2bf(a2.y);
      Bs[nn + 2][kk] = f2bf(a2.z); Bs[nn + 3][kk] = f2bf(a2.w);
    }
    async_wait0();
    __syncthreads();

    #pragma unroll
    for (int ks = 0; ks < KC; ks += 32) {
      ABf a[2], b[2];
      #pragma unroll
      for (int mi = 0; mi < 2; ++mi) {
        int r = wm * 32 + mi * 16 + llo;
        int c0 = ks + (lhi << 4);
        a[mi].q[0] = *(const U4*)&As[r][c0];
        a[mi].q[1] = *(const U4*)&As[r][c0 + 8];
      }
      #pragma unroll
      for (int ni = 0; ni < 2; ++ni) {
        int n = wn * 32 + ni * 16 + llo;
        int kb = ks + (lhi << 4);
        b[ni].q[0] = *(const U4*)&Bs[n][kb];
        b[ni].q[1] = *(const U4*)&Bs[n][kb + 8];
      }
      #pragma unroll
      for (int mi = 0; mi < 2; ++mi)
        #pragma unroll
        for (int ni = 0; ni < 2; ++ni)
          acc[mi][ni] = __builtin_amdgcn_wmma_f32_16x16x32_bf16(
              false, a[mi].v, false, b[ni].v, (short)0, acc[mi][ni], false, false);
    }
    __syncthreads();
  }

  // scatter-accumulate with routing weight (tokens may receive 2 entries)
  #pragma unroll
  for (int mi = 0; mi < 2; ++mi)
    #pragma unroll
    for (int ni = 0; ni < 2; ++ni)
      #pragma unroll
      for (int j = 0; j < 8; ++j) {
        int r = wm * 32 + mi * 16 + (lhi << 3) + j;
        if (m0 + r < cnt) {
          int ge = rowStart + m0 + r;
          int tok = tokenIds[ge];
          float w = wts[ge];
          int col = n0 + wn * 32 + ni * 16 + llo;
          unsafeAtomicAdd(&out[(size_t)tok * HIDDEN + col], w * acc[mi][ni][j]);
        }
      }
}

// ---------------- host launch ----------------

extern "C" void kernel_launch(void* const* d_in, const int* in_sizes, int n_in,
                              void* d_out, int out_size, void* d_ws, size_t ws_size,
                              hipStream_t stream) {
  (void)in_sizes; (void)n_in; (void)out_size; (void)ws_size;
  const float* x    = (const float*)d_in[0];
  const float* ew   = (const float*)d_in[1];
  const float* w1   = (const float*)d_in[2];
  const float* w2   = (const float*)d_in[3];
  const float* w3   = (const float*)d_in[4];
  const int*   eidx = (const int*)d_in[5];
  float* out = (float*)d_out;

  char* ws = (char*)d_ws;
  size_t off = 0;
  auto take = [&](size_t bytes) {
    char* p = ws + off;
    off = (off + bytes + 255) & ~(size_t)255;
    return p;
  };
  unsigned short* xbf  = (unsigned short*)take((size_t)TOKENS * HIDDEN * 2);  // 16 MB
  unsigned short* hbuf = (unsigned short*)take((size_t)NENT * INTER * 2);     // 64 MB
  int*   tokenIds = (int*)take((size_t)NENT * 4);
  float* wts      = (float*)take((size_t)NENT * 4);
  int*   counts   = (int*)take(NUM_EXPERTS * 4);
  int*   offs     = (int*)take((NUM_EXPERTS + 1) * 4);
  int*   cursors  = (int*)take(NUM_EXPERTS * 4);
  int*   zeroPad  = (int*)take(256);

  const int n4 = TOKENS * HIDDEN / 4;
  k_zero_out<<<(n4 + 255) / 256, 256, 0, stream>>>((float4*)out, n4);
  k_init_meta<<<1, 64, 0, stream>>>(counts, zeroPad);
  k_count<<<NENT / 256, 256, 0, stream>>>(eidx, counts);
  k_scan<<<1, 1, 0, stream>>>(counts, offs, cursors);
  k_fill<<<NENT / 256, 256, 0, stream>>>(eidx, ew, cursors, tokenIds, wts);
  k_cvt_x<<<n4 / 256, 256, 0, stream>>>(x, xbf);

  k_gate_up<<<dim3(INTER / TN, NENT / TM, NUM_EXPERTS), NT, 0, stream>>>(
      xbf, w1, w3, tokenIds, offs, (const unsigned short*)zeroPad, hbuf);
  k_down<<<dim3(HIDDEN / TN, NENT / TM, NUM_EXPERTS), NT, 0, stream>>>(
      hbuf, w2, tokenIds, wts, offs, (const unsigned short*)zeroPad, out);
}
